// MaskedAttention_86930138071785
// MI455X (gfx1250) — matrix operand order repaired and checked by reading, not compiled, not run on hardware
//
#include <hip/hip_runtime.h>
#include <hip/hip_fp16.h>

// MI455X (gfx1250) flash-attention for 1x1-conv causal attention.
// Matrix math via v_wmma_f32_16x16x32_f16 (wave32); K/V tiles staged with
// CDNA5 async global->LDS loads (ASYNCcnt) and double buffering.

typedef __attribute__((ext_vector_type(16))) _Float16 v16h;
typedef __attribute__((ext_vector_type(8)))  float    v8f;

union AB16 { v16h v; uint4 q[2]; };

constexpr int N_ = 4, C_ = 256, S_ = 4096, E_ = 256, O_ = 256;
constexpr int OC_ = 768;              // E (q) + E (k) + O (v)
// LDS strides (f16 elems), padded: 16B-aligned rows, distinct banks per row
constexpr int QK_STRIDE = 264;        // 256 + 8
constexpr int VT_STRIDE = 72;         // 64 + 8
constexpr int P_STRIDE  = 72;

// LDS byte offsets for the attention kernel (dynamic smem)
constexpr unsigned QS_OFF  = 0;                       // 64 x 264 f16 = 33792 B
constexpr unsigned KS_OFF  = 33792;                   // 2 x (64 x 264 f16)
constexpr unsigned KS_SZ   = 33792;
constexpr unsigned VT_OFF  = 101376;                  // 2 x (256 x 72 f16)
constexpr unsigned VT_SZ   = 36864;
constexpr unsigned PS_OFF  = 175104;                  // 4 x 16 x 72 f16 = 9216 B
constexpr unsigned SMEM_BYTES = 184320;

__device__ inline v8f wmma16(const AB16& a, const AB16& b, v8f c) {
  return __builtin_amdgcn_wmma_f32_16x16x32_f16(false, a.v, false, b.v,
                                                (short)0, c, false, false);
}

// Per-lane async global->LDS 16B copy (CDNA5 GLOBAL_LOAD_ASYNC_TO_LDS_B128).
__device__ inline void async_b128(const void* gptr, const void* lptr) {
  unsigned la = (unsigned)(uintptr_t)lptr;        // LDS offset = flat[31:0]
  unsigned long long ga = (unsigned long long)(uintptr_t)gptr;
  asm volatile("global_load_async_to_lds_b128 %0, %1, off"
               :: "v"(la), "v"(ga) : "memory");
}

// ---------------- kernel 1: fuse + convert weights to f16 ----------------
__global__ void prep_weights(const float* __restrict__ Wq,
                             const float* __restrict__ Wkv,
                             _Float16* __restrict__ W16) {
  int i = blockIdx.x * 256 + threadIdx.x;
  if (i < OC_ * C_) {
    float w = (i < E_ * C_) ? Wq[i] : Wkv[i - E_ * C_];
    W16[i] = (_Float16)w;
  }
}

// ---------------- kernel 2: WMMA projection --------------------------------
// Block: 256 thr (8 waves) -> tile 32 pixels x 768 out-channels.
__global__ __launch_bounds__(256) void proj_kernel(
    const float* __restrict__ x, const _Float16* __restrict__ W16,
    const float* __restrict__ bq, const float* __restrict__ bkv,
    _Float16* __restrict__ qf, _Float16* __restrict__ kf,
    _Float16* __restrict__ vT) {
  __shared__ _Float16 xs[32 * QK_STRIDE];   // [pixel][channel], transposed stage
  const int n = blockIdx.y;
  const int sBase = blockIdx.x * 32;
  const int t = threadIdx.x;

  { // stage x tile (f32 [C][S] -> f16 LDS [s][c]); coalesced 32-lane reads
    const float* xp = x + (size_t)n * C_ * S_ + sBase;
    const int s = t & 31, c0 = t >> 5;       // c0 in 0..7
    for (int k = 0; k < 32; ++k) {
      int c = c0 + k * 8;
      xs[s * QK_STRIDE + c] = (_Float16)xp[(size_t)c * S_ + s];
    }
  }
  __syncthreads();

  const int w = t >> 5, lane = t & 31;
  const int rg = w >> 2, cg = w & 3;
  const int m = lane & 15, hf = lane >> 4;

  v8f acc[12];
  for (int i = 0; i < 12; ++i)
    for (int r = 0; r < 8; ++r) acc[i][r] = 0.f;

  for (int kc = 0; kc < 8; ++kc) {           // K-depth 256 in chunks of 32
    const int eb = kc * 32;
    AB16 a;                                   // A: 16 pixels x 32 channels
    const _Float16* ap = xs + (rg * 16 + m) * QK_STRIDE + eb + hf * 8;
    a.q[0] = *(const uint4*)ap;
    a.q[1] = *(const uint4*)(ap + 16);
    for (int ti = 0; ti < 12; ++ti) {
      const int oc = cg * 192 + ti * 16 + m;  // B col = output channel
      AB16 b;                                 // B: 32 channels x 16 outputs
      const _Float16* bp = W16 + (size_t)oc * C_ + eb + hf * 8;
      b.q[0] = *(const uint4*)bp;
      b.q[1] = *(const uint4*)(bp + 16);
      acc[ti] = wmma16(a, b, acc[ti]);
    }
  }

  // Epilogue: bias, fold 1/sqrt(E) into q, convert f16, scatter to q/k/vT.
  for (int ti = 0; ti < 12; ++ti) {
    const int gc = cg * 192 + ti * 16 + m;    // uniform matrix per 16-col tile
    const float bv = (gc < E_) ? bq[gc] : bkv[gc - E_];
    const float sc = (gc < E_) ? 0.0625f : 1.0f;   // 1/sqrt(256)
    for (int r = 0; r < 8; ++r) {
      const int s = sBase + rg * 16 + r + 8 * hf;
      const _Float16 hv = (_Float16)((acc[ti][r] + bv) * sc);
      if (gc < E_) {
        qf[((size_t)n * S_ + s) * E_ + gc] = hv;
      } else if (gc < 2 * E_) {
        kf[((size_t)n * S_ + s) * E_ + (gc - E_)] = hv;
      } else {
        vT[((size_t)n * O_ + (gc - 2 * E_)) * S_ + s] = hv;  // v transposed
      }
    }
  }
}

// ---------------- kernel 3: causal flash attention -------------------------
// Block: 128 thr (4 waves), 64 queries/block; wave w owns rows w*16..w*16+15
// and all 256 output columns (16 accumulator tiles). K/V tiles are staged
// with async global->LDS loads, double buffered so the copy of tile kb+1
// overlaps the WMMA work on tile kb.
__global__ __launch_bounds__(128) void attn_kernel(
    const _Float16* __restrict__ qf, const _Float16* __restrict__ kf,
    const _Float16* __restrict__ vT, float* __restrict__ out) {
  extern __shared__ char smem[];
  _Float16* Qs = (_Float16*)(smem + QS_OFF);
  _Float16* Ps = (_Float16*)(smem + PS_OFF);
  float*    Os = (float*)smem;                 // 256 x 65 f32, reuses Qs/Ks0

  const int n = blockIdx.y;
  const int qb = blockIdx.x;
  const int t = threadIdx.x;
  const int w = t >> 5, lane = t & 31;
  const int m = lane & 15, hf = lane >> 4;
  const float NEG_INF = -__builtin_inff();

  // Issue async K/V staging for one 64-key tile (32 b128 instrs per thread).
  auto issue_tile = [&](int kb, int buf) {
    _Float16* KsD = (_Float16*)(smem + KS_OFF + (unsigned)buf * KS_SZ);
    _Float16* VtD = (_Float16*)(smem + VT_OFF + (unsigned)buf * VT_SZ);
    const _Float16* srcK = kf + ((size_t)n * S_ + kb * 64) * E_;
    for (int i = 0; i < 16; ++i) {
      int idx = i * 128 + t;                   // 2048 uint4 per tile
      int row = idx >> 4, c16 = idx & 15;
      async_b128(srcK + idx * 8, KsD + row * QK_STRIDE + c16 * 8);
    }
    for (int i = 0; i < 16; ++i) {
      int idx = i * 128 + t;
      int o = idx >> 3, c16 = idx & 7;
      const _Float16* srcV = vT + ((size_t)n * O_ + o) * S_ + kb * 64;
      async_b128(srcV + c16 * 8, VtD + o * VT_STRIDE + c16 * 8);
    }
  };

  { // stage Q block once (regular loads; independent LDS region)
    const uint4* src = (const uint4*)(qf + ((size_t)n * S_ + qb * 64) * E_);
    for (int i = 0; i < 16; ++i) {
      int idx = i * 128 + t;                   // 2048 uint4
      int row = idx >> 4, c16 = idx & 15;
      *(uint4*)(Qs + row * QK_STRIDE + c16 * 8) = src[idx];
    }
  }
  issue_tile(0, 0);                            // prologue: fill buffer 0

  v8f acc[16];
  for (int i = 0; i < 16; ++i)
    for (int r = 0; r < 8; ++r) acc[i][r] = 0.f;
  float mrow[8], lrow[8];
  for (int r = 0; r < 8; ++r) { mrow[r] = NEG_INF; lrow[r] = 0.f; }

  _Float16* Pw = Ps + w * 16 * P_STRIDE;       // wave-private P tile

  for (int kb = 0; kb <= qb; ++kb) {
    const int cur = kb & 1;
    const _Float16* Ks = (const _Float16*)(smem + KS_OFF + (unsigned)cur * KS_SZ);
    const _Float16* Vt = (const _Float16*)(smem + VT_OFF + (unsigned)cur * VT_SZ);

    // (A) all waves finished reading buf[cur^1] (tile kb-1) -> safe to refill
    __syncthreads();
    const bool hasNext = (kb + 1 <= qb);
    if (hasNext) issue_tile(kb + 1, cur ^ 1);
    // Wait for the current tile's 32 async loads (in-order completion); the
    // next tile's 32 may still be in flight.
    if (hasNext) asm volatile("s_wait_asynccnt 32" ::: "memory");
    else         asm volatile("s_wait_asynccnt 0"  ::: "memory");
    // (B) every wave's share of the current tile has landed in LDS
    __syncthreads();

    // ---- S = Q K^T : 16 rows x 64 keys per wave (q pre-scaled by 1/sqrtE)
    v8f sv[4];
    for (int j = 0; j < 4; ++j)
      for (int r = 0; r < 8; ++r) sv[j][r] = 0.f;
    for (int ec = 0; ec < 8; ++ec) {
      const int eb = ec * 32;
      AB16 a;
      const _Float16* ap = Qs + (w * 16 + m) * QK_STRIDE + eb + hf * 8;
      a.q[0] = *(const uint4*)ap;
      a.q[1] = *(const uint4*)(ap + 16);
      for (int j = 0; j < 4; ++j) {
        AB16 b;                                // B col = key j*16+m
        const _Float16* bp = Ks + (j * 16 + m) * QK_STRIDE + eb + hf * 8;
        b.q[0] = *(const uint4*)bp;
        b.q[1] = *(const uint4*)(bp + 16);
        sv[j] = wmma16(a, b, sv[j]);
      }
    }

    // ---- causal mask (only the diagonal block needs it)
    if (kb == qb) {
      for (int j = 0; j < 4; ++j) {
        const int key = kb * 64 + j * 16 + m;
        for (int r = 0; r < 8; ++r) {
          const int row = qb * 64 + w * 16 + r + 8 * hf;
          if (key > row) sv[j][r] = NEG_INF;
        }
      }
    }

    // ---- online softmax. Row r+8*hf lives in VGPR slot r of lane-half hf;
    // xor-shuffles 1,2,4,8 reduce within each 16-lane half.
    float mnew[8], alpha[8];
    for (int r = 0; r < 8; ++r) {
      float v = fmaxf(fmaxf(sv[0][r], sv[1][r]), fmaxf(sv[2][r], sv[3][r]));
      v = fmaxf(v, __shfl_xor(v, 1));
      v = fmaxf(v, __shfl_xor(v, 2));
      v = fmaxf(v, __shfl_xor(v, 4));
      v = fmaxf(v, __shfl_xor(v, 8));
      mnew[r] = fmaxf(mrow[r], v);
      alpha[r] = __expf(mrow[r] - mnew[r]);    // first iter: exp(-inf)=0
    }
    for (int r = 0; r < 8; ++r) {
      float rowsum = 0.f;
      for (int j = 0; j < 4; ++j) {
        float p = __expf(sv[j][r] - mnew[r]);
        sv[j][r] = p;
        rowsum += p;
      }
      rowsum += __shfl_xor(rowsum, 1);
      rowsum += __shfl_xor(rowsum, 2);
      rowsum += __shfl_xor(rowsum, 4);
      rowsum += __shfl_xor(rowsum, 8);
      lrow[r] = lrow[r] * alpha[r] + rowsum;
      mrow[r] = mnew[r];
    }
    for (int ti = 0; ti < 16; ++ti)
      for (int r = 0; r < 8; ++r) acc[ti][r] *= alpha[r];

    // ---- D-layout -> A-layout for P via wave-private LDS tile
    for (int j = 0; j < 4; ++j)
      for (int r = 0; r < 8; ++r)
        Pw[(r + 8 * hf) * P_STRIDE + j * 16 + m] = (_Float16)sv[j][r];

    AB16 pa[2];
    for (int c = 0; c < 2; ++c) {
      const _Float16* ap = Pw + m * P_STRIDE + c * 32 + hf * 8;
      pa[c].q[0] = *(const uint4*)ap;
      pa[c].q[1] = *(const uint4*)(ap + 16);
    }
    // ---- acc += P @ V  (B from transposed Vt: contiguous along key depth)
    for (int ti = 0; ti < 16; ++ti) {
      for (int c = 0; c < 2; ++c) {
        AB16 b;
        const _Float16* bp = Vt + (ti * 16 + m) * VT_STRIDE + c * 32 + hf * 8;
        b.q[0] = *(const uint4*)bp;
        b.q[1] = *(const uint4*)(bp + 16);
        acc[ti] = wmma16(pa[c], b, acc[ti]);
      }
    }
  }

  // ---- epilogue: normalize, transpose through LDS, coalesced [n][o][s] store
  __syncthreads();
  for (int ti = 0; ti < 16; ++ti) {
    const int o = ti * 16 + m;
    for (int r = 0; r < 8; ++r) {
      const int sl = w * 16 + r + 8 * hf;
      Os[o * 65 + sl] = acc[ti][r] / lrow[r];
    }
  }
  __syncthreads();
  float* outp = out + (size_t)n * O_ * S_ + qb * 64;
  for (int i = 0; i < 128; ++i) {
    int idx = i * 128 + t;                     // 256*64 f32
    int o = idx >> 6, sl = idx & 63;
    outp[(size_t)o * S_ + sl] = Os[o * 65 + sl];
  }
}

// ---------------- launch ---------------------------------------------------
extern "C" void kernel_launch(void* const* d_in, const int* in_sizes, int n_in,
                              void* d_out, int out_size, void* d_ws,
                              size_t ws_size, hipStream_t stream) {
  const float* x   = (const float*)d_in[0];
  const float* Wq  = (const float*)d_in[1];
  const float* bq  = (const float*)d_in[2];
  const float* Wkv = (const float*)d_in[3];
  const float* bkv = (const float*)d_in[4];
  float* out = (float*)d_out;

  char* ws = (char*)d_ws;
  _Float16* qf  = (_Float16*)ws;                          // 8 MiB
  _Float16* kf  = (_Float16*)(ws + ((size_t)8 << 20));    // 8 MiB
  _Float16* vT  = (_Float16*)(ws + ((size_t)16 << 20));   // 8 MiB
  _Float16* W16 = (_Float16*)(ws + ((size_t)24 << 20));   // 384 KiB

  prep_weights<<<(OC_ * C_ + 255) / 256, 256, 0, stream>>>(Wq, Wkv, W16);

  dim3 pg(S_ / 32, N_);
  proj_kernel<<<pg, 256, 0, stream>>>(x, W16, bq, bkv, qf, kf, vT);

  dim3 ag(S_ / 64, N_);
  attn_kernel<<<ag, 128, SMEM_BYTES, stream>>>(qf, kf, vT, out);
}
